// GraphGuidedMambaBlock_22076131902163
// MI455X (gfx1250) — matrix operand order repaired and checked by reading, not compiled, not run on hardware
//
#include <hip/hip_runtime.h>
#include <cstddef>

// ---------------- problem constants (from reference) ----------------
#define B_SZ    8
#define T_LEN   64
#define N_NODES 325
#define D_MODEL 64
#define D_INNER 128
#define D_STATE 16
#define DT_RANK 4
#define TOPK    8
#define HEADS   4
#define HD      16
#define CHUNK   16       // T-chunk per Mamba GEMM pass

typedef __bf16 bf16;
typedef __attribute__((ext_vector_type(16))) __bf16 v16bf;
typedef __attribute__((ext_vector_type(8)))  float  v8f;

// ---------------- WMMA helpers (CDNA5 gfx1250, wave32) ----------------
__device__ __forceinline__ v8f wmma_bf16(v16bf a, v16bf b, v8f c) {
  // (neg_a, A, neg_b, B, c_mod, C, reuse_a, reuse_b)
  return __builtin_amdgcn_wmma_f32_16x16x32_bf16(false, a, false, b, (short)0, c,
                                                 false, false);
}

// A fragment: 16x32 (MxK) slice from an f32 row-major tile, bf16-converted.
// Lane L (r=L&15, hi=L>>4): K = {k0+hi*8 .. +7} and {k0+16+hi*8 .. +7} of row r.
__device__ __forceinline__ v16bf load_fragA_f32(const float* A, int lda, int row0,
                                                int k0) {
  int lane = threadIdx.x & 31;
  const float* p = A + (size_t)(row0 + (lane & 15)) * lda + k0 + ((lane >> 4) << 3);
  v16bf f;
#pragma unroll
  for (int i = 0; i < 8; ++i) { f[i] = (bf16)p[i]; f[i + 8] = (bf16)p[i + 16]; }
  return f;
}

// B fragment: 32x16 (KxN) slice where B = W^T, W is (N x K) row-major f32.
// Lane L (n=L&15, hi=L>>4): K = {k0+hi*8 .. +7} and {k0+16+hi*8 .. +7} of W row (row0+n).
// Rows >= nrows yield zeros (used to pad x_proj's 36 rows up to 48).
__device__ __forceinline__ v16bf load_fragB_f32(const float* W, int ldw, int row0,
                                                int k0, int nrows) {
  int lane = threadIdx.x & 31;
  int r = row0 + (lane & 15);
  v16bf f;
  if (r < nrows) {
    const float* p = W + (size_t)r * ldw + k0 + ((lane >> 4) << 3);
#pragma unroll
    for (int i = 0; i < 8; ++i) { f[i] = (bf16)p[i]; f[i + 8] = (bf16)p[i + 16]; }
  } else {
#pragma unroll
    for (int i = 0; i < 16; ++i) f[i] = (bf16)0.0f;
  }
  return f;
}

__device__ __forceinline__ float sigmoidf_(float x) { return 1.0f / (1.0f + expf(-x)); }
__device__ __forceinline__ float siluf_(float x)    { return x * sigmoidf_(x); }
__device__ __forceinline__ float softplusf_(float x) {
  return (x > 20.0f) ? x : log1pf(expf(x));
}

// =====================================================================
// Kernel 1: per-sequence fused LN1 + Mamba (+ residual) -> h1 (f32)
//   grid.x = B*N (2600), block = 256 (8 waves). T processed in 16-row chunks,
//   all intermediates in LDS; SSM scan state kept in registers (lane = channel).
// =====================================================================
__global__ __launch_bounds__(256) void mamba_kernel(
    const float* __restrict__ x, const float* __restrict__ ln1_g,
    const float* __restrict__ ln1_b, const float* __restrict__ in_proj_w,
    const float* __restrict__ conv_w, const float* __restrict__ conv_b,
    const float* __restrict__ x_proj_w, const float* __restrict__ dt_w,
    const float* __restrict__ dt_b, const float* __restrict__ A_log,
    const float* __restrict__ D_ssm, const float* __restrict__ out_w,
    float* __restrict__ h1) {
  __shared__ float s_x[CHUNK][D_MODEL];        // raw chunk (residual)
  __shared__ float s_ln[CHUNK][D_MODEL + 4];   // LN output (padded stride)
  __shared__ float s_xi[CHUNK + 3][D_INNER];   // conv input, rows 0..2 = history
  __shared__ float s_z[CHUNK][D_INNER];        // gate branch
  __shared__ float s_xc[CHUNK][D_INNER + 4];   // conv+silu output
  __shared__ float s_dbl[CHUNK][48];           // x_proj output (dt|B|C, padded)
  __shared__ float s_dt[CHUNK][D_INNER];       // softplus(dt)
  __shared__ float s_y[CHUNK][D_INNER + 4];    // scan output * silu(z)

  const int seq  = blockIdx.x;
  const int b    = seq / N_NODES;
  const int n    = seq % N_NODES;
  const int tid  = threadIdx.x;
  const int wave = tid >> 5;
  const int lane = tid & 31;
  const int nn   = lane & 15;
  const int m0   = (lane >> 4) << 3;

  // per-channel scan state (threads 0..127 own channel d = tid)
  float hs[D_STATE], Areg[D_STATE], Dv = 0.0f;
  if (tid < D_INNER) {
#pragma unroll
    for (int s = 0; s < D_STATE; ++s) {
      Areg[s] = -expf(A_log[tid * D_STATE + s]);
      hs[s] = 0.0f;
    }
    Dv = D_ssm[tid];
    s_xi[0][tid] = 0.0f; s_xi[1][tid] = 0.0f; s_xi[2][tid] = 0.0f;  // conv pad
  }
  __syncthreads();

  for (int c = 0; c < T_LEN / CHUNK; ++c) {
    // ---- carry conv history from previous chunk (per-column, no conflicts)
    if (c > 0 && tid < D_INNER) {
      s_xi[0][tid] = s_xi[CHUNK + 0][tid];
      s_xi[1][tid] = s_xi[CHUNK + 1][tid];
      s_xi[2][tid] = s_xi[CHUNK + 2][tid];
    }
    // ---- load raw chunk, coalesced
#pragma unroll
    for (int k = 0; k < (CHUNK * D_MODEL) / 256; ++k) {
      int idx = tid + k * 256;
      int i = idx >> 6, d = idx & 63;
      s_x[i][d] = x[((size_t)(b * T_LEN + c * CHUNK + i) * N_NODES + n) * D_MODEL + d];
    }
    __syncthreads();

    // ---- LayerNorm (16 rows, one thread each; trivially cheap)
    if (tid < CHUNK) {
      float m = 0.0f;
#pragma unroll
      for (int d = 0; d < D_MODEL; ++d) m += s_x[tid][d];
      m *= (1.0f / D_MODEL);
      float v = 0.0f;
#pragma unroll
      for (int d = 0; d < D_MODEL; ++d) { float t = s_x[tid][d] - m; v += t * t; }
      v *= (1.0f / D_MODEL);
      float rs = rsqrtf(v + 1e-5f);
#pragma unroll
      for (int d = 0; d < D_MODEL; ++d)
        s_ln[tid][d] = (s_x[tid][d] - m) * rs * ln1_g[d] + ln1_b[d];
    }
    __syncthreads();

    // ---- GEMM1: xz(16x256) = ln(16x64) @ in_proj_w^T.  8 waves x 2 N-tiles.
    {
      v8f acc0 = {0,0,0,0,0,0,0,0}, acc1 = {0,0,0,0,0,0,0,0};
#pragma unroll
      for (int kk = 0; kk < D_MODEL; kk += 32) {
        v16bf a  = load_fragA_f32(&s_ln[0][0], D_MODEL + 4, 0, kk);
        v16bf b0 = load_fragB_f32(in_proj_w, D_MODEL, wave * 16, kk, 256);
        v16bf b1 = load_fragB_f32(in_proj_w, D_MODEL, (wave + 8) * 16, kk, 256);
        acc0 = wmma_bf16(a, b0, acc0);
        acc1 = wmma_bf16(a, b1, acc1);
      }
      int col = wave * 16 + nn;   // tile 'wave' -> xi cols, tile 'wave+8' -> z cols
#pragma unroll
      for (int v = 0; v < 8; ++v) {
        int row = m0 + v;
        s_xi[3 + row][col] = acc0[v];
        s_z[row][col]      = acc1[v];
      }
    }
    __syncthreads();

    // ---- depthwise conv (width 4, causal) + SiLU
#pragma unroll
    for (int k = 0; k < (CHUNK * D_INNER) / 256; ++k) {
      int idx = tid + k * 256;
      int i = idx >> 7, d = idx & 127;
      float acc = conv_b[d];
#pragma unroll
      for (int j = 0; j < 4; ++j) acc += s_xi[i + j][d] * conv_w[d * 4 + j];
      s_xc[i][d] = siluf_(acc);
    }
    __syncthreads();

    // ---- GEMM2: dbl(16x36) = xc(16x128) @ x_proj_w^T (padded to 48 cols)
    if (wave < 3) {
      v8f acc = {0,0,0,0,0,0,0,0};
#pragma unroll
      for (int kk = 0; kk < D_INNER; kk += 32) {
        v16bf a = load_fragA_f32(&s_xc[0][0], D_INNER + 4, 0, kk);
        v16bf bb = load_fragB_f32(x_proj_w, D_INNER, wave * 16, kk,
                                  DT_RANK + 2 * D_STATE);
        acc = wmma_bf16(a, bb, acc);
      }
#pragma unroll
      for (int v = 0; v < 8; ++v) s_dbl[m0 + v][wave * 16 + nn] = acc[v];
    }
    __syncthreads();

    // ---- dt projection + softplus (rank 4 -> 128)
#pragma unroll
    for (int k = 0; k < (CHUNK * D_INNER) / 256; ++k) {
      int idx = tid + k * 256;
      int i = idx >> 7, d = idx & 127;
      float a = dt_b[d];
#pragma unroll
      for (int r = 0; r < DT_RANK; ++r) a += s_dbl[i][r] * dt_w[d * DT_RANK + r];
      s_dt[i][d] = softplusf_(a);
    }
    __syncthreads();

    // ---- selective scan: lane = channel d, 16-state recurrence in registers
    if (tid < D_INNER) {
      const int d = tid;
      for (int i = 0; i < CHUNK; ++i) {
        float dtv = s_dt[i][d];
        float xv  = s_xc[i][d];
        float dtx = dtv * xv;
        float y   = 0.0f;
#pragma unroll
        for (int s = 0; s < D_STATE; ++s) {
          float dA = expf(dtv * Areg[s]);
          hs[s] = hs[s] * dA + dtx * s_dbl[i][DT_RANK + s];
          y += hs[s] * s_dbl[i][DT_RANK + D_STATE + s];
        }
        float zv = s_z[i][d];
        s_y[i][d] = (y + Dv * xv) * siluf_(zv);
      }
    }
    __syncthreads();

    // ---- GEMM3: out(16x64) = y(16x128) @ out_w^T ; add residual; write h1
    if (wave < 4) {
      v8f acc = {0,0,0,0,0,0,0,0};
#pragma unroll
      for (int kk = 0; kk < D_INNER; kk += 32) {
        v16bf a = load_fragA_f32(&s_y[0][0], D_INNER + 4, 0, kk);
        v16bf bb = load_fragB_f32(out_w, D_INNER, wave * 16, kk, D_MODEL);
        acc = wmma_bf16(a, bb, acc);
      }
      int col = wave * 16 + nn;
#pragma unroll
      for (int v = 0; v < 8; ++v) {
        int i = m0 + v;
        size_t off =
            ((size_t)(b * T_LEN + c * CHUNK + i) * N_NODES + n) * D_MODEL + col;
        h1[off] = s_x[i][col] + acc[v];
      }
    }
    __syncthreads();
  }
}

// =====================================================================
// Kernel 2: K/V projections, h1(Rx64) @ {k,v}_w^T + bias -> bf16 workspace.
//   grid = (R/64, 2), block = 256. R = B*T*N = 166400 (divisible by 64).
// =====================================================================
__global__ __launch_bounds__(256) void kv_proj_kernel(
    const float* __restrict__ h1, const float* __restrict__ k_w,
    const float* __restrict__ k_b, const float* __restrict__ v_w,
    const float* __restrict__ v_b, unsigned short* __restrict__ Kb_,
    unsigned short* __restrict__ Vb_) {
  __shared__ float s_h[64][D_MODEL + 4];
  const int r0   = blockIdx.x * 64;
  const int sel  = blockIdx.y;
  const float* W    = sel ? v_w : k_w;
  const float* bias = sel ? v_b : k_b;
  bf16* dst = reinterpret_cast<bf16*>(sel ? Vb_ : Kb_);
  const int tid = threadIdx.x, wave = tid >> 5;
  const int lane = tid & 31, nn = lane & 15, m0 = (lane >> 4) << 3;

#pragma unroll
  for (int k = 0; k < 16; ++k) {
    int idx = tid + k * 256;
    int i = idx >> 6, d = idx & 63;
    s_h[i][d] = h1[(size_t)(r0 + i) * D_MODEL + d];
  }
  __syncthreads();

  for (int tile = wave * 2; tile < wave * 2 + 2; ++tile) {   // 16 tiles / 8 waves
    int mt = tile >> 2, nt = tile & 3;
    v8f acc = {0,0,0,0,0,0,0,0};
#pragma unroll
    for (int kk = 0; kk < D_MODEL; kk += 32) {
      v16bf a = load_fragA_f32(&s_h[0][0], D_MODEL + 4, mt * 16, kk);
      v16bf bb = load_fragB_f32(W, D_MODEL, nt * 16, kk, D_MODEL);
      acc = wmma_bf16(a, bb, acc);
    }
    int col = nt * 16 + nn;
#pragma unroll
    for (int v = 0; v < 8; ++v) {
      int row = mt * 16 + m0 + v;
      dst[(size_t)(r0 + row) * D_MODEL + col] = (bf16)(acc[v] + bias[col]);
    }
  }
}

// =====================================================================
// Kernel 3: per-64-row tile: Q proj (WMMA) -> top-8 neighbor attention ->
//           o-proj (WMMA) -> gating MLP -> blend -> LayerNorm2 -> d_out.
// =====================================================================
__global__ __launch_bounds__(256) void attn_out_kernel(
    const float* __restrict__ h1, const unsigned short* __restrict__ Kb_,
    const unsigned short* __restrict__ Vb_, const int* __restrict__ nbr,
    const float* __restrict__ q_w, const float* __restrict__ q_b,
    const float* __restrict__ o_w, const float* __restrict__ o_b,
    const float* __restrict__ g1_w, const float* __restrict__ g1_b,
    const float* __restrict__ g2_w, const float* __restrict__ g2_b,
    const float* __restrict__ ln2_g, const float* __restrict__ ln2_b,
    float* __restrict__ out) {
  __shared__ float s_h[64][D_MODEL + 4];
  __shared__ float s_qog[64][D_MODEL + 4];   // Q, then reused for attention output
  __shared__ float s_xg[64][D_MODEL];
  const bf16* Kb = reinterpret_cast<const bf16*>(Kb_);
  const bf16* Vb = reinterpret_cast<const bf16*>(Vb_);
  const int r0 = blockIdx.x * 64;
  const int tid = threadIdx.x, wave = tid >> 5;
  const int lane = tid & 31, nn = lane & 15, m0 = (lane >> 4) << 3;

#pragma unroll
  for (int k = 0; k < 16; ++k) {
    int idx = tid + k * 256;
    int i = idx >> 6, d = idx & 63;
    s_h[i][d] = h1[(size_t)(r0 + i) * D_MODEL + d];
  }
  __syncthreads();

  // ---- Q = h1_tile @ q_w^T + q_b
  for (int tile = wave * 2; tile < wave * 2 + 2; ++tile) {
    int mt = tile >> 2, nt = tile & 3;
    v8f acc = {0,0,0,0,0,0,0,0};
#pragma unroll
    for (int kk = 0; kk < D_MODEL; kk += 32) {
      v16bf a = load_fragA_f32(&s_h[0][0], D_MODEL + 4, mt * 16, kk);
      v16bf bb = load_fragB_f32(q_w, D_MODEL, nt * 16, kk, D_MODEL);
      acc = wmma_bf16(a, bb, acc);
    }
    int col = nt * 16 + nn;
#pragma unroll
    for (int v = 0; v < 8; ++v)
      s_qog[mt * 16 + m0 + v][col] = acc[v] + q_b[col];
  }
  __syncthreads();

  // ---- attention: one thread per (row-in-tile, head)
  {
    const int i = tid >> 2, hh = tid & 3;
    float q[HD];
#pragma unroll
    for (int j = 0; j < HD; ++j) q[j] = s_qog[i][hh * HD + j];
    __syncthreads();   // Q fully read; s_qog becomes attention-output buffer

    const int r = r0 + i;
    const int n = r % N_NODES;
    const int bt = r / N_NODES;   // b*T + t
    float logit[TOPK];
    int   base[TOPK];
    float mx = -3.4e38f;
#pragma unroll
    for (int k = 0; k < TOPK; ++k) {
      int nb = nbr[n * TOPK + k];
      int rk = (bt * N_NODES + nb) * D_MODEL + hh * HD;
      base[k] = rk;
      float dot = 0.0f;
#pragma unroll
      for (int j = 0; j < HD; ++j) dot += q[j] * (float)Kb[rk + j];
      logit[k] = dot * 0.25f;                 // hd^-0.5, hd=16
      mx = fmaxf(mx, logit[k]);
    }
    float se = 0.0f;
#pragma unroll
    for (int k = 0; k < TOPK; ++k) { logit[k] = expf(logit[k] - mx); se += logit[k]; }
    float inv = 1.0f / se;
    float og[HD];
#pragma unroll
    for (int j = 0; j < HD; ++j) og[j] = 0.0f;
#pragma unroll
    for (int k = 0; k < TOPK; ++k) {
      float w = logit[k] * inv;
#pragma unroll
      for (int j = 0; j < HD; ++j) og[j] += w * (float)Vb[base[k] + j];
    }
#pragma unroll
    for (int j = 0; j < HD; ++j) s_qog[i][hh * HD + j] = og[j];
  }
  __syncthreads();

  // ---- x_graph = og @ o_w^T + o_b
  for (int tile = wave * 2; tile < wave * 2 + 2; ++tile) {
    int mt = tile >> 2, nt = tile & 3;
    v8f acc = {0,0,0,0,0,0,0,0};
#pragma unroll
    for (int kk = 0; kk < D_MODEL; kk += 32) {
      v16bf a = load_fragA_f32(&s_qog[0][0], D_MODEL + 4, mt * 16, kk);
      v16bf bb = load_fragB_f32(o_w, D_MODEL, nt * 16, kk, D_MODEL);
      acc = wmma_bf16(a, bb, acc);
    }
    int col = nt * 16 + nn;
#pragma unroll
    for (int v = 0; v < 8; ++v) s_xg[mt * 16 + m0 + v][col] = acc[v] + o_b[col];
  }
  __syncthreads();

  // ---- gate MLP + blend + LayerNorm2, one thread per row
  if (tid < 64) {
    const int i = tid;
    float g2acc = g2_b[0];
#pragma unroll
    for (int cI = 0; cI < 16; ++cI) {
      float u = g1_b[cI];
#pragma unroll
      for (int d = 0; d < D_MODEL; ++d) u += s_h[i][d] * g1_w[cI * D_MODEL + d];
      float gl = 0.5f * u * (1.0f + erff(u * 0.70710678118f));  // exact gelu
      g2acc += gl * g2_w[cI];
    }
    float g = sigmoidf_(g2acc);
    float m = 0.0f;
#pragma unroll
    for (int d = 0; d < D_MODEL; ++d) {
      float val = s_h[i][d] + g * (s_xg[i][d] - s_h[i][d]);
      s_xg[i][d] = val;
      m += val;
    }
    m *= (1.0f / D_MODEL);
    float v = 0.0f;
#pragma unroll
    for (int d = 0; d < D_MODEL; ++d) { float t = s_xg[i][d] - m; v += t * t; }
    v *= (1.0f / D_MODEL);
    float rs = rsqrtf(v + 1e-5f);
    size_t obase = (size_t)(r0 + i) * D_MODEL;
#pragma unroll
    for (int d = 0; d < D_MODEL; ++d)
      out[obase + d] = (s_xg[i][d] - m) * rs * ln2_g[d] + ln2_b[d];
  }
}

// =====================================================================
extern "C" void kernel_launch(void* const* d_in, const int* in_sizes, int n_in,
                              void* d_out, int out_size, void* d_ws, size_t ws_size,
                              hipStream_t stream) {
  (void)in_sizes; (void)n_in; (void)out_size; (void)ws_size;
  const float* x        = (const float*)d_in[0];
  const int*   nbr      = (const int*)  d_in[1];
  const float* ln1_g    = (const float*)d_in[2];
  const float* ln1_b    = (const float*)d_in[3];
  const float* in_proj_w= (const float*)d_in[4];
  const float* conv_w   = (const float*)d_in[5];
  const float* conv_b   = (const float*)d_in[6];
  const float* x_proj_w = (const float*)d_in[7];
  const float* dt_w     = (const float*)d_in[8];
  const float* dt_b     = (const float*)d_in[9];
  const float* A_log    = (const float*)d_in[10];
  const float* D_ssm    = (const float*)d_in[11];
  const float* out_w    = (const float*)d_in[12];
  const float* q_w      = (const float*)d_in[13];
  const float* q_b      = (const float*)d_in[14];
  const float* k_w      = (const float*)d_in[15];
  const float* k_b      = (const float*)d_in[16];
  const float* v_w      = (const float*)d_in[17];
  const float* v_b      = (const float*)d_in[18];
  const float* o_w      = (const float*)d_in[19];
  const float* o_b      = (const float*)d_in[20];
  const float* g1_w     = (const float*)d_in[21];
  const float* g1_b     = (const float*)d_in[22];
  const float* g2_w     = (const float*)d_in[23];
  const float* g2_b     = (const float*)d_in[24];
  const float* ln2_g    = (const float*)d_in[25];
  const float* ln2_b    = (const float*)d_in[26];

  const size_t R = (size_t)B_SZ * T_LEN * N_NODES;   // 166400 rows (divisible by 64)
  // workspace layout: h1 f32 (R*64), then K,V as bf16 (R*64 each) ~ 85 MB total
  float*          h1 = (float*)d_ws;
  unsigned short* Kb = (unsigned short*)((char*)d_ws + R * D_MODEL * sizeof(float));
  unsigned short* Vb = Kb + R * D_MODEL;

  mamba_kernel<<<B_SZ * N_NODES, 256, 0, stream>>>(
      x, ln1_g, ln1_b, in_proj_w, conv_w, conv_b, x_proj_w, dt_w, dt_b, A_log,
      D_ssm, out_w, h1);

  dim3 gkv((unsigned)(R / 64), 2);
  kv_proj_kernel<<<gkv, 256, 0, stream>>>(h1, k_w, k_b, v_w, v_b, Kb, Vb);

  attn_out_kernel<<<(unsigned)(R / 64), 256, 0, stream>>>(
      h1, Kb, Vb, nbr, q_w, q_b, o_w, o_b, g1_w, g1_b, g2_w, g2_b, ln2_g, ln2_b,
      (float*)d_out);
}
